// LigerLMHeadKTO_74698071212406
// MI455X (gfx1250) — matrix operand order repaired and checked by reading, not compiled, not run on hardware
//
#include <hip/hip_runtime.h>
#include <hip/hip_bf16.h>
#include <stdint.h>

// ---------------------------------------------------------------------------
// KTO LM-head loss for MI455X (gfx1250).
//  - BF16 WMMA (v_wmma_f32_16x16x32_bf16), fp32 accumulate
//  - V-slice-parallel flash-logsumexp (W streamed from HBM exactly once)
//  - 256KB X token-tiles staged in LDS via the Tensor Data Mover
//  - XOR-swizzled X layout -> conflict-free ds_load_b128 A fragments
//  - double-buffered W fragment loads hide global latency under WMMA
// ---------------------------------------------------------------------------

typedef __attribute__((ext_vector_type(16))) __bf16       v16bf;
typedef __attribute__((ext_vector_type(8)))  float        v8f;
typedef __attribute__((ext_vector_type(4)))  unsigned int u32x4;
typedef __attribute__((ext_vector_type(8)))  int          i32x8;
typedef __attribute__((ext_vector_type(4)))  int          i32x4;

#define KTO_IGNORE (-100)
#define KTO_BETA   0.1f

constexpr int BSZ    = 4;
constexpr int SEQT   = 512;
constexpr int TOK    = 2048;            // BSZ*SEQT
constexpr int HDIM   = 2048;
constexpr int VOC    = 32000;

constexpr int NBLK   = 128;             // vocab columns per block
constexpr int SLICES = VOC / NBLK;      // 250
constexpr int MTILE  = 64;              // tokens per LDS tile
constexpr int MTILES = TOK / MTILE;     // 32

// workspace layout (bytes)
constexpr size_t XB_OFF  = 0;
constexpr size_t XB_SZ   = (size_t)2 * TOK * HDIM * 2;        // bf16 X (swizzled)
constexpr size_t PM_OFF  = XB_OFF + XB_SZ;
constexpr size_t P_SZ    = (size_t)2 * SLICES * TOK * 4;
constexpr size_t PS_OFF  = PM_OFF + P_SZ;
constexpr size_t TGT_OFF = PS_OFF + P_SZ;
constexpr size_t TGT_SZ  = (size_t)2 * TOK * 4;
constexpr size_t TL_OFF  = TGT_OFF + TGT_SZ;

constexpr int LDS_A_BYTES = MTILE * HDIM * 2;                      // 262144
constexpr int LDS_TOTAL   = LDS_A_BYTES + 4 * MTILE * 4 * 2 + MTILE * 4;

// ---------------------------------------------------------------------------
// Kernel 0: fp32 -> bf16 + XOR swizzle. One 16-byte output block (8 elems)
// per thread; block index within a row is XORed with (row & 15) so the GEMM's
// ds_load_b128 A-fragment reads are bank-conflict-free (row stride 4096 B
// would otherwise alias all 16 lanes onto the same banks).
// ---------------------------------------------------------------------------
__global__ void kto_cvt(const float* __restrict__ x, const float* __restrict__ rx,
                        __bf16* __restrict__ xb)
{
    size_t i    = (size_t)blockIdx.x * blockDim.x + threadIdx.x;  // 8-elem group id
    size_t half = (size_t)TOK * HDIM / 8;
    const float4* src = (i < half) ? (const float4*)x : (const float4*)rx;
    size_t j = (i < half) ? i : (i - half);
    float4 f0 = src[j * 2];
    float4 f1 = src[j * 2 + 1];
    union { __bf16 h[8]; uint4 u; } o;
    o.h[0] = (__bf16)f0.x; o.h[1] = (__bf16)f0.y;
    o.h[2] = (__bf16)f0.z; o.h[3] = (__bf16)f0.w;
    o.h[4] = (__bf16)f1.x; o.h[5] = (__bf16)f1.y;
    o.h[6] = (__bf16)f1.z; o.h[7] = (__bf16)f1.w;
    size_t e    = i * 8;                 // element idx in [0, 2*TOK*HDIM)
    size_t row  = e / HDIM;              // global row (TOK % 16 == 0, so &15 ok)
    size_t kblk = (e % HDIM) >> 3;       // 16-byte block within row
    size_t ksw  = kblk ^ (row & 15);
    ((uint4*)xb)[row * (HDIM / 8) + ksw] = o.u;
}

// ---- helpers --------------------------------------------------------------
__device__ inline void loadw(const float* __restrict__ p0, const float* __restrict__ p1,
                             int k, int g, float4* d0, float4* d1)
{
    const float4* q0 = (const float4*)(p0 + k + g * 8);
    const float4* q1 = (const float4*)(p1 + k + g * 8);
    d0[0] = q0[0]; d0[1] = q0[1]; d0[2] = q0[4]; d0[3] = q0[5];
    d1[0] = q1[0]; d1[1] = q1[1]; d1[2] = q1[4]; d1[3] = q1[5];
}

__device__ inline v16bf cvtregs(const float4* f)
{
    v16bf r;
    r[0]=(__bf16)f[0].x;  r[1]=(__bf16)f[0].y;  r[2]=(__bf16)f[0].z;  r[3]=(__bf16)f[0].w;
    r[4]=(__bf16)f[1].x;  r[5]=(__bf16)f[1].y;  r[6]=(__bf16)f[1].z;  r[7]=(__bf16)f[1].w;
    r[8]=(__bf16)f[2].x;  r[9]=(__bf16)f[2].y;  r[10]=(__bf16)f[2].z; r[11]=(__bf16)f[2].w;
    r[12]=(__bf16)f[3].x; r[13]=(__bf16)f[3].y; r[14]=(__bf16)f[3].z; r[15]=(__bf16)f[3].w;
    return r;
}

// A fragment: two swizzled 16B K-runs per lane (ISA 16-bit A layout)
__device__ inline v16bf loada(const char* abase, int k0, int g, int ln)
{
    union { v16bf v; uint4 q[2]; } a;
    int blk = (k0 >> 3) + g;
    a.q[0] = *(const uint4*)(abase + ((size_t)((blk)     ^ ln) << 4));
    a.q[1] = *(const uint4*)(abase + ((size_t)((blk + 2) ^ ln) << 4));
    return a.v;
}

// ---------------------------------------------------------------------------
// Kernel 1: per-(model, V-slice) GEMM + flash partial logsumexp.
// grid = (SLICES, 2); block = 256 (8 waves, 2(M) x 4(N), 32x32 C per wave).
// ---------------------------------------------------------------------------
__global__ void kto_gemm_partial(const float* __restrict__ W0,
                                 const float* __restrict__ W1,
                                 const __bf16* __restrict__ XB,
                                 const int* __restrict__ y,
                                 float* __restrict__ pm,
                                 float* __restrict__ ps,
                                 float* __restrict__ tgt)
{
    extern __shared__ char smem[];                       // [0,262144): X tile
    float* redm = (float*)(smem + LDS_A_BYTES);          // [4][MTILE]
    float* reds = redm + 4 * MTILE;                      // [4][MTILE]
    int*   yt   = (int*)(reds + 4 * MTILE);              // [MTILE]

    const int slice = blockIdx.x;
    const int mdl   = blockIdx.y;
    const float*  Wm = mdl ? W1 : W0;
    const __bf16* Xm = XB + (size_t)mdl * TOK * HDIM;
    const int v0 = slice * NBLK;

    const int tid  = threadIdx.x;
    const int w    = tid >> 5;
    const int lane = tid & 31;
    const int wm   = w >> 2;        // 0..1 : M quadrant (32 rows)
    const int wn   = w & 3;         // 0..3 : N quadrant (32 cols)
    const int g    = lane >> 4;     // 16-lane half
    const int ln   = lane & 15;

    const char* abase0 = smem + (size_t)(wm * 32 + ln)      * (HDIM * 2);
    const char* abase1 = smem + (size_t)(wm * 32 + 16 + ln) * (HDIM * 2);
    const float* wrow0 = Wm + (size_t)(v0 + wn * 32 + ln)      * HDIM;
    const float* wrow1 = Wm + (size_t)(v0 + wn * 32 + 16 + ln) * HDIM;

    for (int mt = 0; mt < MTILES; ++mt) {
        // ---- TDM: DMA 64x2048 bf16 X tile into LDS (wave 0 issues; EXEC ignored)
        if (w == 0) {
            uint64_t ga = (uint64_t)(uintptr_t)(Xm + (size_t)mt * MTILE * HDIM);
            u32x4 g0;
            g0[0] = 1u;                                       // count=1, user load
            g0[1] = 0u;                                       // lds_addr = 0
            g0[2] = (unsigned)(ga & 0xffffffffu);             // global_addr[31:0]
            g0[3] = (unsigned)((ga >> 32) & 0x01ffffffu) | (2u << 30); // [56:32]|type=2
            i32x8 g1;
            g1[0] = (int)(1u << 16);                          // data_size=2B, mask=0
            g1[1] = (int)(((unsigned)HDIM & 0xffffu) << 16);  // tensor_dim0 lo16
            g1[2] = (int)((((unsigned)HDIM >> 16) & 0xffffu)  // tensor_dim0 hi16
                        | (((unsigned)MTILE & 0xffffu) << 16));// tensor_dim1 lo16
            g1[3] = (int)(((unsigned)HDIM & 0xffffu) << 16);  // tile_dim0 = 2048
            g1[4] = (int)MTILE;                               // tile_dim1 = 64
            g1[5] = (int)HDIM;                                // dim0_stride lo32
            g1[6] = 0;
            g1[7] = 0;
            i32x4 gz = {0, 0, 0, 0};
#if defined(__clang_major__) && (__clang_major__ >= 23)
            i32x8 gz8 = {0, 0, 0, 0, 0, 0, 0, 0};
            __builtin_amdgcn_tensor_load_to_lds(g0, g1, gz, gz, gz8, 0);
#else
            __builtin_amdgcn_tensor_load_to_lds(g0, g1, gz, gz, 0);
#endif
            __builtin_amdgcn_s_wait_tensorcnt(0);
        }
        if (tid < MTILE) yt[tid] = y[mt * MTILE + tid];   // stage labels once
        __syncthreads();

        // ---- GEMM: C[64 x 128], K swept over H, W loads double-buffered
        v8f c00 = {}, c01 = {}, c10 = {}, c11 = {};       // [mh][ch] 16x16 tiles
        float4 wa0[4], wa1[4], wb0[4], wb1[4];
        loadw(wrow0, wrow1, 0, g, wa0, wa1);
        for (int k0 = 0; k0 < HDIM; k0 += 64) {
            // stage k0+32 while computing k0
            loadw(wrow0, wrow1, k0 + 32, g, wb0, wb1);
            {
                v16bf a0 = loada(abase0, k0, g, ln);
                v16bf a1 = loada(abase1, k0, g, ln);
                v16bf b0 = cvtregs(wa0);
                v16bf b1 = cvtregs(wa1);
                c00 = __builtin_amdgcn_wmma_f32_16x16x32_bf16(false, a0, false, b0, (short)0, c00, false, false);
                c01 = __builtin_amdgcn_wmma_f32_16x16x32_bf16(false, a0, false, b1, (short)0, c01, false, false);
                c10 = __builtin_amdgcn_wmma_f32_16x16x32_bf16(false, a1, false, b0, (short)0, c10, false, false);
                c11 = __builtin_amdgcn_wmma_f32_16x16x32_bf16(false, a1, false, b1, (short)0, c11, false, false);
            }
            // stage k0+64 (dummy k=0 reload on the last iteration) while computing k0+32
            int kn = (k0 + 64 < HDIM) ? (k0 + 64) : 0;
            loadw(wrow0, wrow1, kn, g, wa0, wa1);
            {
                v16bf a0 = loada(abase0, k0 + 32, g, ln);
                v16bf a1 = loada(abase1, k0 + 32, g, ln);
                v16bf b0 = cvtregs(wb0);
                v16bf b1 = cvtregs(wb1);
                c00 = __builtin_amdgcn_wmma_f32_16x16x32_bf16(false, a0, false, b0, (short)0, c00, false, false);
                c01 = __builtin_amdgcn_wmma_f32_16x16x32_bf16(false, a0, false, b1, (short)0, c01, false, false);
                c10 = __builtin_amdgcn_wmma_f32_16x16x32_bf16(false, a1, false, b0, (short)0, c10, false, false);
                c11 = __builtin_amdgcn_wmma_f32_16x16x32_bf16(false, a1, false, b1, (short)0, c11, false, false);
            }
        }

        // ---- per-token max over this wave's 32 cols + target-logit capture
        float mloc[2][8];
        for (int mh = 0; mh < 2; ++mh) {
            for (int r = 0; r < 8; ++r) {
                int tloc  = wm * 32 + mh * 16 + r + g * 8;
                int tglob = mt * MTILE + tloc;
                float e0 = mh ? c10[r] : c00[r];
                float e1 = mh ? c11[r] : c01[r];
                int lab  = yt[tloc];
                int col0 = v0 + wn * 32 + ln;
                if (lab == col0)      tgt[(size_t)mdl * TOK + tglob] = e0;
                if (lab == col0 + 16) tgt[(size_t)mdl * TOK + tglob] = e1;
                float v = fmaxf(e0, e1);
                v = fmaxf(v, __shfl_xor(v, 1, 16));
                v = fmaxf(v, __shfl_xor(v, 2, 16));
                v = fmaxf(v, __shfl_xor(v, 4, 16));
                v = fmaxf(v, __shfl_xor(v, 8, 16));
                mloc[mh][r] = v;
                if (ln == 0) redm[wn * MTILE + tloc] = v;
            }
        }
        __syncthreads();

        // ---- block max over 128 cols, then exp-sum
        for (int mh = 0; mh < 2; ++mh) {
            for (int r = 0; r < 8; ++r) {
                int tloc = wm * 32 + mh * 16 + r + g * 8;
                float mb = fmaxf(fmaxf(redm[0 * MTILE + tloc], redm[1 * MTILE + tloc]),
                                 fmaxf(redm[2 * MTILE + tloc], redm[3 * MTILE + tloc]));
                mloc[mh][r] = mb;
                float e0 = mh ? c10[r] : c00[r];
                float e1 = mh ? c11[r] : c01[r];
                float s = __expf(e0 - mb) + __expf(e1 - mb);
                s += __shfl_xor(s, 1, 16);
                s += __shfl_xor(s, 2, 16);
                s += __shfl_xor(s, 4, 16);
                s += __shfl_xor(s, 8, 16);
                if (ln == 0) reds[wn * MTILE + tloc] = s;
            }
        }
        __syncthreads();

        // ---- one writer per token: slice partial (m, s)
        if (wn == 0 && ln == 0) {
            for (int mh = 0; mh < 2; ++mh) {
                for (int r = 0; r < 8; ++r) {
                    int tloc  = wm * 32 + mh * 16 + r + g * 8;
                    int tglob = mt * MTILE + tloc;
                    float sb = reds[0 * MTILE + tloc] + reds[1 * MTILE + tloc]
                             + reds[2 * MTILE + tloc] + reds[3 * MTILE + tloc];
                    size_t o = ((size_t)mdl * SLICES + slice) * TOK + tglob;
                    pm[o] = mloc[mh][r];
                    ps[o] = sb;
                }
            }
        }
        __syncthreads();   // LDS tile reusable for next mt
    }
}

// ---------------------------------------------------------------------------
// Kernel 2: merge slice partials -> per-token log-prob (online logsumexp)
// ---------------------------------------------------------------------------
__global__ void kto_token_reduce(const float* __restrict__ pm, const float* __restrict__ ps,
                                 const float* __restrict__ tgt, const int* __restrict__ y,
                                 float* __restrict__ tl)
{
    int idx = blockIdx.x * blockDim.x + threadIdx.x;     // 0 .. 2*TOK-1
    if (idx >= 2 * TOK) return;
    int mdl = idx / TOK, tok = idx % TOK;
    const float* pmb = pm + (size_t)mdl * SLICES * TOK + tok;
    const float* psb = ps + (size_t)mdl * SLICES * TOK + tok;
    float m = -3.4e38f, s = 0.f;
    for (int sl = 0; sl < SLICES; ++sl) {
        float mm = pmb[(size_t)sl * TOK];
        float ss = psb[(size_t)sl * TOK];
        float nm = fmaxf(m, mm);
        s = s * __expf(m - nm) + ss * __expf(mm - nm);
        m = nm;
    }
    float logZ = m + __logf(s);
    int lab = y[tok];
    tl[idx] = (lab == KTO_IGNORE) ? 0.f : (tgt[idx] - logZ);
}

// ---------------------------------------------------------------------------
// Kernel 3: masked per-sequence means + KTO loss (single block)
// ---------------------------------------------------------------------------
__global__ void kto_final(const float* __restrict__ tl, const int* __restrict__ y,
                          const int* __restrict__ pref, float* __restrict__ out)
{
    __shared__ float av[2][BSZ];
    int w = threadIdx.x >> 5, lane = threadIdx.x & 31;
    int mdl = w >> 2, seq = w & 3;                       // 8 warps = 2 models x 4 seqs
    float sum = 0.f; float cnt = 0.f;
    for (int i = lane; i < SEQT; i += 32) {
        int t = seq * SEQT + i;
        sum += tl[mdl * TOK + t];
        cnt += (y[t] != KTO_IGNORE) ? 1.f : 0.f;
    }
    for (int mk = 16; mk >= 1; mk >>= 1) {
        sum += __shfl_xor(sum, mk, 32);
        cnt += __shfl_xor(cnt, mk, 32);
    }
    if (lane == 0) av[mdl][seq] = sum / cnt;
    __syncthreads();
    if (threadIdx.x == 0) {
        float loss = 0.f;
        for (int b = 0; b < BSZ; ++b) {
            float lr   = av[0][b] - av[1][b];
            float mult = pref[b] ? 1.f : -1.f;
            float z    = KTO_BETA * lr * mult;
            loss += 1.f - 1.f / (1.f + __expf(-z));
        }
        out[0] = loss / (float)BSZ;
    }
}

// ---------------------------------------------------------------------------
extern "C" void kernel_launch(void* const* d_in, const int* in_sizes, int n_in,
                              void* d_out, int out_size, void* d_ws, size_t ws_size,
                              hipStream_t stream)
{
    const float* x    = (const float*)d_in[0];
    const float* rx   = (const float*)d_in[1];
    const int*   y    = (const int*)d_in[2];
    const int*   pref = (const int*)d_in[3];
    const float* W    = (const float*)d_in[4];
    const float* rW   = (const float*)d_in[5];

    char* ws = (char*)d_ws;
    __bf16* xb  = (__bf16*)(ws + XB_OFF);
    float*  pm  = (float*)(ws + PM_OFF);
    float*  psum= (float*)(ws + PS_OFF);
    float*  tgt = (float*)(ws + TGT_OFF);
    float*  tl  = (float*)(ws + TL_OFF);

    // 0) fp32 -> bf16 (+bank swizzle) for x / ref_x; 8 elems per thread
    kto_cvt<<<4096, 256, 0, stream>>>(x, rx, xb);
    // 1) GEMM + slice-partial logsumexp: 250 V-slices x 2 models
    kto_gemm_partial<<<dim3(SLICES, 2), 256, LDS_TOTAL, stream>>>(W, rW, xb, y, pm, psum, tgt);
    // 2) per-token logsumexp merge
    kto_token_reduce<<<16, 256, 0, stream>>>(pm, psum, tgt, y, tl);
    // 3) masked sequence means + KTO loss
    kto_final<<<1, 256, 0, stream>>>(tl, y, pref, (float*)d_out);
}
